// TENProLayer_7017976561863
// MI455X (gfx1250) — compile-verified, matrix operand order checked by reading, not run on hardware
//
#include <hip/hip_runtime.h>
#include <hip/hip_bf16.h>

// ---------------------------------------------------------------- types
typedef __bf16 bf16_t;
typedef __attribute__((ext_vector_type(16))) __bf16 v16bf;
typedef __attribute__((ext_vector_type(8)))  __bf16 v8bf;
typedef __attribute__((ext_vector_type(8)))  float  v8f;
typedef int v4i_ __attribute__((vector_size(16)));   // matches builtin param type

#define AS1 __attribute__((address_space(1)))
#define AS3 __attribute__((address_space(3)))

// problem constants (match reference setup_inputs)
#define BB   4
#define TT   4096
#define DD   1024
#define KK   256
#define HH   8
#define KPH  32
#define MLPD 4096
#define NT   (BB * TT)        // 16384 tokens
#define CH   128              // scan chunk length
#define NC   (TT / CH)        // 32 chunks

// GEMM tiling
#define BM   128
#define BN   128
#define KC   32               // K per stage (one WMMA k-step)
#define LDT  40               // LDS row pitch in halves (20 dwords: bank-conflict free)

// ---------------- async copy to LDS (gfx1250), guarded with sync fallback
#if defined(__gfx1250__) && __has_builtin(__builtin_amdgcn_global_load_async_to_lds_b128)
#define HAVE_ASYNC_LDS 1
__device__ __forceinline__ void async_copy16(const void* g, void* l) {
    __builtin_amdgcn_global_load_async_to_lds_b128(
        (AS1 v4i_*)(uintptr_t)g, (AS3 v4i_*)(uint32_t)(uintptr_t)l, 0, 0);
}
#else
#define HAVE_ASYNC_LDS 0
__device__ __forceinline__ void async_copy16(const void* g, void* l) {
    *(v8bf*)l = *(const v8bf*)g;   // sync fallback: global load + ds store
}
#endif

#if __has_builtin(__builtin_amdgcn_s_wait_asynccnt)
#define WAIT_ASYNC(n) __builtin_amdgcn_s_wait_asynccnt(n)
#else
#define WAIT_ASYNC(n) asm volatile("s_wait_asynccnt %0" ::"i"(n) : "memory")
#endif

__device__ __forceinline__ void wait_stage(bool more_inflight) {
#if HAVE_ASYNC_LDS
    if (more_inflight) { WAIT_ASYNC(4); } else { WAIT_ASYNC(0); }
#else
    (void)more_inflight;
#endif
}

// ---------------------------------------------------------------- helpers
__device__ __forceinline__ float sigmoidf_(float x) { return 1.0f / (1.0f + __expf(-x)); }

// Read one 16x32 bf16 WMMA fragment from an LDS tile (row pitch LDT halves).
// lanes 0-15: rows 0-15, K = 0..7 (halves 0-7) and 16..23 (halves 8-15)
// lanes 16-31: rows 0-15, K = 8..15 and 24..31
__device__ __forceinline__ v16bf lds_frag(const bf16_t* __restrict__ tile,
                                          int row0, int lane) {
    const int r  = row0 + (lane & 15);
    const int kb = (lane >> 4) << 3;
    const bf16_t* p = tile + r * LDT + kb;
    union { v16bf v; v8bf h[2]; } u;
    u.h[0] = *(const v8bf*)(p);
    u.h[1] = *(const v8bf*)(p + 16);
    return u.v;
}

// ---------------------------------------------------------------- WMMA GEMM
// C[M,N] (f32 acc) = A[M,Kd](bf16 row-major) * Bt[N,Kd](bf16 K-contiguous).
// 256 threads = 8 waves; block tile 128x128; wave tile 32(M) x 64(N);
// LDS double-buffered stages filled with async global->LDS b128 copies.
// MODE epilogues:
//  0: outF = acc
//  1: outF += sigmoid(*scale_ptr) * acc            (eigenstate memory mix)
//  2: outF = sigmoid(acc + bias[n])                (spectral gate)
//  3: outF = aux2 + aux * acc                      (x1 = residual + gate*h)
//  4: outH = bf16(silu(acc + bias[n]))             (MLP hidden)
//  5: outF = acc + bias[n] + aux                   (x_out = mlp2 + b2 + x1)
template <int MODE>
__global__ __launch_bounds__(256)
void gemm_bf16_wmma(const bf16_t* __restrict__ A, const bf16_t* __restrict__ Bt,
                    float* __restrict__ outF, bf16_t* __restrict__ outH,
                    const float* __restrict__ bias, const float* __restrict__ aux,
                    const float* __restrict__ aux2, const float* __restrict__ scale_ptr,
                    int M, int N, int Kd)
{
    __shared__ __align__(16) bf16_t sA[2][BM * LDT];
    __shared__ __align__(16) bf16_t sB[2][BN * LDT];

    const int tid  = threadIdx.x;
    const int lane = tid & 31;
    const int wid  = tid >> 5;
    const int wm   = wid & 3;          // 4 waves in M -> 32 rows each
    const int wn   = wid >> 2;         // 2 waves in N -> 64 cols each
    const int bm0  = blockIdx.y * BM;
    const int bn0  = blockIdx.x * BN;

    // stage copy: 128 rows x 32 cols per matrix; 512 16B-chunks / matrix;
    // 256 threads -> 2 chunks each for A, 2 for B (4 async b128 per thread).
    const int c0r = tid >> 2, c0c = (tid & 3) << 3;          // chunk tid
    const int c1r = (tid + 256) >> 2, c1c = c0c;             // chunk tid+256

    auto copy_stage = [&](int buf, int k0) {
        async_copy16(A + (size_t)(bm0 + c0r) * Kd + k0 + c0c, &sA[buf][c0r * LDT + c0c]);
        async_copy16(A + (size_t)(bm0 + c1r) * Kd + k0 + c1c, &sA[buf][c1r * LDT + c1c]);
        async_copy16(Bt + (size_t)(bn0 + c0r) * Kd + k0 + c0c, &sB[buf][c0r * LDT + c0c]);
        async_copy16(Bt + (size_t)(bn0 + c1r) * Kd + k0 + c1c, &sB[buf][c1r * LDT + c1c]);
    };

    v8f acc[2][4];
#pragma unroll
    for (int mi = 0; mi < 2; ++mi)
#pragma unroll
        for (int j = 0; j < 4; ++j)
#pragma unroll
            for (int e = 0; e < 8; ++e) acc[mi][j][e] = 0.0f;

    const int nk = Kd / KC;
    copy_stage(0, 0);
    if (nk > 1) copy_stage(1, KC);

    for (int i = 0; i < nk; ++i) {
        wait_stage(i + 1 < nk);        // this wave's stage-i copies are done
        __syncthreads();               // everyone's stage-i copies are done
        const int buf = i & 1;

        v16bf a0 = lds_frag(sA[buf], wm * 32, lane);
        v16bf a1 = lds_frag(sA[buf], wm * 32 + 16, lane);
#pragma unroll
        for (int j = 0; j < 4; ++j) {
            v16bf b = lds_frag(sB[buf], wn * 64 + j * 16, lane);
            acc[0][j] = __builtin_amdgcn_wmma_f32_16x16x32_bf16(
                false, a0, false, b, (short)0, acc[0][j], false, false);
            acc[1][j] = __builtin_amdgcn_wmma_f32_16x16x32_bf16(
                false, a1, false, b, (short)0, acc[1][j], false, false);
        }
        __syncthreads();               // all reads of buf done before refill
        if (i + 2 < nk) copy_stage(buf, (i + 2) * KC);
    }

    float sc = 0.0f;
    if (MODE == 1) sc = sigmoidf_(*scale_ptr);

    const int mloc = (lane >> 4) << 3;   // 0 or 8
    const int nloc = lane & 15;
#pragma unroll
    for (int mi = 0; mi < 2; ++mi) {
#pragma unroll
        for (int j = 0; j < 4; ++j) {
            const int col = bn0 + wn * 64 + j * 16 + nloc;
#pragma unroll
            for (int r = 0; r < 8; ++r) {
                const int row = bm0 + wm * 32 + mi * 16 + mloc + r;
                const size_t idx = (size_t)row * N + col;
                const float v = acc[mi][j][r];
                if (MODE == 0) {
                    outF[idx] = v;
                } else if (MODE == 1) {
                    outF[idx] += sc * v;
                } else if (MODE == 2) {
                    outF[idx] = sigmoidf_(v + bias[col]);
                } else if (MODE == 3) {
                    outF[idx] = aux2[idx] + aux[idx] * v;
                } else if (MODE == 4) {
                    float t = v + bias[col];
                    outH[idx] = (bf16_t)(t * sigmoidf_(t));   // silu
                } else if (MODE == 5) {
                    outF[idx] = v + bias[col] + aux[idx];
                }
            }
        }
    }
}

// ---------------------------------------------------------------- layernorm
__global__ __launch_bounds__(256)
void layernorm_bf16(const float* __restrict__ x, const float* __restrict__ g,
                    const float* __restrict__ b, bf16_t* __restrict__ out, int D)
{
    const int row = blockIdx.x;
    const float* xr = x + (size_t)row * D;
    float s = 0.0f, s2 = 0.0f;
    for (int i = threadIdx.x; i < D; i += 256) {
        float v = xr[i];
        s += v; s2 += v * v;
    }
#pragma unroll
    for (int o = 16; o > 0; o >>= 1) {
        s  += __shfl_down(s,  o, 32);
        s2 += __shfl_down(s2, o, 32);
    }
    __shared__ float ps[8], ps2[8];
    const int lane = threadIdx.x & 31, w = threadIdx.x >> 5;
    if (lane == 0) { ps[w] = s; ps2[w] = s2; }
    __syncthreads();
    if (threadIdx.x == 0) {
        float a = 0.0f, a2 = 0.0f;
        for (int i = 0; i < 8; ++i) { a += ps[i]; a2 += ps2[i]; }
        ps[0] = a; ps2[0] = a2;
    }
    __syncthreads();
    const float mean = ps[0] / D;
    const float var  = ps2[0] / D - mean * mean;
    const float inv  = rsqrtf(var + 1e-5f);
    bf16_t* orow = out + (size_t)row * D;
    for (int i = threadIdx.x; i < D; i += 256)
        orow[i] = (bf16_t)((xr[i] - mean) * inv * g[i] + b[i]);
}

// ---------------------------------------------------------------- converts
__global__ void cvt_f32_to_bf16(const float* __restrict__ s, bf16_t* __restrict__ d, size_t n)
{
    size_t i = (size_t)blockIdx.x * blockDim.x + threadIdx.x;
    if (i < n) d[i] = (bf16_t)s[i];
}

// dst[c*R + r] = bf16(src[r*C + c])   (store K-contiguous weight columns)
__global__ __launch_bounds__(256)
void transpose_f32_to_bf16(const float* __restrict__ src, bf16_t* __restrict__ dst,
                           int R, int C)
{
    __shared__ float tile[32][33];
    const int cb = blockIdx.x * 32, rb = blockIdx.y * 32;
    const int tx = threadIdx.x & 31, ty = threadIdx.x >> 5;  // 32 x 8
#pragma unroll
    for (int i = 0; i < 32; i += 8)
        tile[ty + i][tx] = src[(size_t)(rb + ty + i) * C + cb + tx];
    __syncthreads();
#pragma unroll
    for (int i = 0; i < 32; i += 8)
        dst[(size_t)(cb + ty + i) * R + rb + tx] = (bf16_t)tile[tx][ty + i];
}

// ---------------------------------------------------------------- chunked scan
// c(t) = lambda_k * c(t-1) + beta(t),  lambda = sigmoid(log_decay)*e^{i*freq}
__global__ __launch_bounds__(256)
void scan_local_kernel(const float* __restrict__ beta, const float* __restrict__ logdec,
                       const float* __restrict__ freq, float* __restrict__ cr,
                       float* __restrict__ ci, float2* __restrict__ carry)
{
    const int tid   = blockIdx.x * blockDim.x + threadIdx.x;   // B*NC*K threads
    const int k     = tid & (KK - 1);
    const int chunk = (tid >> 8) & (NC - 1);
    const int b     = tid >> 13;          // / (KK*NC) = /8192
    const float mag = sigmoidf_(logdec[k]);
    const float fr  = freq[k];
    const float lr = mag * __cosf(fr), li = mag * __sinf(fr);
    float sr = 0.0f, si = 0.0f;
    const size_t tok0 = (size_t)b * TT + (size_t)chunk * CH;
    for (int t = 0; t < CH; ++t) {
        const size_t tok = tok0 + t;
        const float br = beta[tok * (2 * KK) + k];
        const float bi = beta[tok * (2 * KK) + KK + k];
        const float nr = lr * sr - li * si + br;
        const float ni = lr * si + li * sr + bi;
        sr = nr; si = ni;
        cr[tok * KK + k] = sr;
        ci[tok * KK + k] = si;
    }
    carry[((size_t)b * NC + chunk) * KK + k] = make_float2(sr, si);
}

__global__ __launch_bounds__(256)
void scan_carry_kernel(const float* __restrict__ logdec, const float* __restrict__ freq,
                       const float2* __restrict__ carry, float2* __restrict__ sbuf)
{
    const int tid = blockIdx.x * blockDim.x + threadIdx.x;    // B*K threads
    const int k = tid & (KK - 1);
    const int b = tid >> 8;
    const float mag = sigmoidf_(logdec[k]);
    const float mp  = __powf(mag, (float)CH);
    float sn, cs;
    __sincosf(freq[k] * (float)CH, &sn, &cs);
    const float Lr = mp * cs, Li = mp * sn;       // lambda^CH
    float sr = 0.0f, si = 0.0f;                   // s_0 = 0
    for (int j = 0; j < NC; ++j) {
        sbuf[((size_t)b * NC + j) * KK + k] = make_float2(sr, si);
        const float2 e = carry[((size_t)b * NC + j) * KK + k];
        const float nr = Lr * sr - Li * si + e.x; // s_{j+1} = e_j + L*s_j
        const float ni = Lr * si + Li * sr + e.y;
        sr = nr; si = ni;
    }
}

__global__ __launch_bounds__(256)
void scan_fixup_kernel(const float* __restrict__ logdec, const float* __restrict__ freq,
                       const float2* __restrict__ sbuf, float* __restrict__ cr,
                       float* __restrict__ ci)
{
    const int tid   = blockIdx.x * blockDim.x + threadIdx.x;   // B*NC*K threads
    const int k     = tid & (KK - 1);
    const int chunk = (tid >> 8) & (NC - 1);
    const int b     = tid >> 13;
    const float2 s = sbuf[((size_t)b * NC + chunk) * KK + k];
    const float mag = sigmoidf_(logdec[k]);
    const float fr  = freq[k];
    const float lr = mag * __cosf(fr), li = mag * __sinf(fr);
    float pr = lr, pi = li;                        // lambda^(tl+1), tl=0
    const size_t tok0 = (size_t)b * TT + (size_t)chunk * CH;
    for (int t = 0; t < CH; ++t) {
        const size_t tok = tok0 + t;
        cr[tok * KK + k] += pr * s.x - pi * s.y;
        ci[tok * KK + k] += pr * s.y + pi * s.x;
        const float nr = pr * lr - pi * li;
        const float ni = pr * li + pi * lr;
        pr = nr; pi = ni;
    }
}

// ---------------------------------------------------------------- coupling
// out[b,t,h,j] = sum_k coupling[h,j,k] * c[b,t,h,k]; writes eigenstates
// [B,T,2K] both f32 (second model output) and bf16 (for out_proj WMMA).
__global__ __launch_bounds__(256)
void coupling_kernel(const float* __restrict__ cr, const float* __restrict__ ci,
                     const float* __restrict__ coup, float* __restrict__ eigF,
                     bf16_t* __restrict__ eigH)
{
    const int tid = blockIdx.x * blockDim.x + threadIdx.x;  // NT*256 threads
    const int j = tid & 31;
    const int h = (tid >> 5) & 7;
    const size_t tok = (size_t)(tid >> 8);
    const float* cw = coup + ((size_t)h * KPH + j) * KPH;
    const size_t cb = tok * KK + h * KPH;
    float ar = 0.0f, ai = 0.0f;
#pragma unroll
    for (int k = 0; k < KPH; ++k) {
        const float w = cw[k];
        ar += w * cr[cb + k];
        ai += w * ci[cb + k];
    }
    const size_t eo = tok * (2 * KK) + h * KPH + j;
    eigF[eo]      = ar;
    eigF[eo + KK] = ai;
    eigH[eo]      = (bf16_t)ar;
    eigH[eo + KK] = (bf16_t)ai;
}

// ---------------------------------------------------------------- launch
extern "C" void kernel_launch(void* const* d_in, const int* in_sizes, int n_in,
                              void* d_out, int out_size, void* d_ws, size_t ws_size,
                              hipStream_t stream)
{
    const float* x      = (const float*)d_in[0];   // [B,T,D]
    const float* prev   = (const float*)d_in[1];   // [B,T,2K]
    const float* w_in   = (const float*)d_in[2];   // [D,2K]
    const float* logdec = (const float*)d_in[3];   // [K]
    const float* freq   = (const float*)d_in[4];   // [K]
    const float* mgate  = (const float*)d_in[5];   // scalar
    const float* w_mem  = (const float*)d_in[6];   // [2K,2K] (used as W.T -> no transpose)
    const float* coup   = (const float*)d_in[7];   // [H,KPH,KPH]
    const float* w_gate = (const float*)d_in[8];   // [D,D]
    const float* b_gate = (const float*)d_in[9];   // [D]
    const float* w_out  = (const float*)d_in[10];  // [2K,D]
    const float* w_m1   = (const float*)d_in[11];  // [D,MLP]
    const float* b_m1   = (const float*)d_in[12];  // [MLP]
    const float* w_m2   = (const float*)d_in[13];  // [MLP,D]
    const float* b_m2   = (const float*)d_in[14];  // [D]
    const float* n1g = (const float*)d_in[15];
    const float* n1b = (const float*)d_in[16];
    const float* n2g = (const float*)d_in[17];
    const float* n2b = (const float*)d_in[18];

    float* xout_f = (float*)d_out;                       // [B,T,D]
    float* eig_f  = (float*)d_out + (size_t)NT * DD;     // [B,T,2K]

    // ---------------- workspace carve-out
    char* base = (char*)d_ws;
    size_t off = 0;
    auto take = [&](size_t bytes) -> void* {
        void* p = base + off;
        off = (off + bytes + 255) & ~(size_t)255;
        return p;
    };
    bf16_t* xnorm_h = (bf16_t*)take((size_t)NT * DD * 2);
    bf16_t* prev_h  = (bf16_t*)take((size_t)NT * 2 * KK * 2);
    float*  betaF   = (float*) take((size_t)NT * 2 * KK * 4);
    float*  c_r     = (float*) take((size_t)NT * KK * 4);
    float*  c_i     = (float*) take((size_t)NT * KK * 4);
    float2* carry   = (float2*)take((size_t)BB * NC * KK * 8);
    float2* sbuf    = (float2*)take((size_t)BB * NC * KK * 8);
    bf16_t* eig_h   = (bf16_t*)take((size_t)NT * 2 * KK * 2);
    float*  gateF   = (float*) take((size_t)NT * DD * 4);
    float*  x1F     = (float*) take((size_t)NT * DD * 4);
    bf16_t* x2n_h   = (bf16_t*)take((size_t)NT * DD * 2);
    bf16_t* hid_h   = (bf16_t*)take((size_t)NT * MLPD * 2);
    bf16_t* w_inT   = (bf16_t*)take((size_t)2 * KK * DD * 2);   // [512,1024]
    bf16_t* w_memB  = (bf16_t*)take((size_t)2 * KK * 2 * KK * 2);
    bf16_t* w_gateT = (bf16_t*)take((size_t)DD * DD * 2);
    bf16_t* w_outT  = (bf16_t*)take((size_t)DD * 2 * KK * 2);   // [1024,512]
    bf16_t* w_m1T   = (bf16_t*)take((size_t)MLPD * DD * 2);     // [4096,1024]
    bf16_t* w_m2T   = (bf16_t*)take((size_t)DD * MLPD * 2);     // [1024,4096]

    // ---------------- weight prep (bf16, K-contiguous per output column)
    transpose_f32_to_bf16<<<dim3(2 * KK / 32, DD / 32), 256, 0, stream>>>(w_in,   w_inT,   DD,     2 * KK);
    transpose_f32_to_bf16<<<dim3(DD / 32,     DD / 32), 256, 0, stream>>>(w_gate, w_gateT, DD,     DD);
    transpose_f32_to_bf16<<<dim3(DD / 32, 2 * KK / 32), 256, 0, stream>>>(w_out,  w_outT,  2 * KK, DD);
    transpose_f32_to_bf16<<<dim3(MLPD / 32,   DD / 32), 256, 0, stream>>>(w_m1,   w_m1T,   DD,     MLPD);
    transpose_f32_to_bf16<<<dim3(DD / 32,   MLPD / 32), 256, 0, stream>>>(w_m2,   w_m2T,   MLPD,   DD);
    // y = prev @ W.T  => Bt[n][k] = W[n][k]: straight convert, already K-contiguous
    {
        size_t n = (size_t)2 * KK * 2 * KK;
        cvt_f32_to_bf16<<<(unsigned)((n + 255) / 256), 256, 0, stream>>>(w_mem, w_memB, n);
    }
    {
        size_t n = (size_t)NT * 2 * KK;
        cvt_f32_to_bf16<<<(unsigned)((n + 255) / 256), 256, 0, stream>>>(prev, prev_h, n);
    }

    // ---------------- 1) LN1
    layernorm_bf16<<<NT, 256, 0, stream>>>(x, n1g, n1b, xnorm_h, DD);

    // ---------------- 2) beta = x_norm @ in_proj  (+ mg * prev @ Wmem.T)
    gemm_bf16_wmma<0><<<dim3(2 * KK / BN, NT / BM), 256, 0, stream>>>(
        xnorm_h, w_inT, betaF, nullptr, nullptr, nullptr, nullptr, nullptr, NT, 2 * KK, DD);
    gemm_bf16_wmma<1><<<dim3(2 * KK / BN, NT / BM), 256, 0, stream>>>(
        prev_h, w_memB, betaF, nullptr, nullptr, nullptr, nullptr, mgate, NT, 2 * KK, 2 * KK);

    // ---------------- 3) chunked complex scan (replaces the FFT conv)
    scan_local_kernel<<<(BB * NC * KK) / 256, 256, 0, stream>>>(betaF, logdec, freq, c_r, c_i, carry);
    scan_carry_kernel<<<(BB * KK) / 256, 256, 0, stream>>>(logdec, freq, carry, sbuf);
    scan_fixup_kernel<<<(BB * NC * KK) / 256, 256, 0, stream>>>(logdec, freq, sbuf, c_r, c_i);

    // ---------------- 4) per-head coupling -> eigenstates (f32 out + bf16)
    coupling_kernel<<<(NT * 256) / 256, 256, 0, stream>>>(c_r, c_i, coup, eig_f, eig_h);

    // ---------------- 5) gate = sigmoid(x_norm @ gate_w + gate_b)
    gemm_bf16_wmma<2><<<dim3(DD / BN, NT / BM), 256, 0, stream>>>(
        xnorm_h, w_gateT, gateF, nullptr, b_gate, nullptr, nullptr, nullptr, NT, DD, DD);

    // ---------------- 6) x1 = x + gate * (eigenstates @ out_proj)
    gemm_bf16_wmma<3><<<dim3(DD / BN, NT / BM), 256, 0, stream>>>(
        eig_h, w_outT, x1F, nullptr, nullptr, gateF, x, nullptr, NT, DD, 2 * KK);

    // ---------------- 7) LN2 + MLP
    layernorm_bf16<<<NT, 256, 0, stream>>>(x1F, n2g, n2b, x2n_h, DD);
    gemm_bf16_wmma<4><<<dim3(MLPD / BN, NT / BM), 256, 0, stream>>>(
        x2n_h, w_m1T, nullptr, hid_h, b_m1, nullptr, nullptr, nullptr, NT, MLPD, DD);
    gemm_bf16_wmma<5><<<dim3(DD / BN, NT / BM), 256, 0, stream>>>(
        hid_h, w_m2T, xout_f, nullptr, b_m2, x1F, nullptr, nullptr, NT, DD, MLPD);

    (void)in_sizes; (void)n_in; (void)out_size; (void)ws_size;
}
